// ParaphraseClassifierExtra_63333587746927
// MI455X (gfx1250) — compile-verified, matrix operand order checked by reading, not run on hardware
//
#include <hip/hip_runtime.h>
#include <hip/hip_bf16.h>

// ---------------------------------------------------------------------------
// Problem constants (from reference)
// ---------------------------------------------------------------------------
#define SEQ   60
#define BATCH 256
#define EMB   300     // D
#define DPAD  320     // D padded to multiple of 64 (K dim)
#define NPAD  304     // D padded to multiple of 16 (N dim of proj output)
#define HID   512     // H
#define G4H   2048    // 4*H
#define SPAD  64      // SEQ padded to multiple of 16
#define DM    1024
#define MIN_  232     // MLP_IN
#define MINP  256     // MLP_IN padded

typedef unsigned short u16;
typedef unsigned int   u32;

typedef __attribute__((ext_vector_type(16))) __bf16 v16bf;
typedef __attribute__((ext_vector_type(8)))  float  v8f;

union BFrag { v16bf v; u32 u[8]; };

__device__ __forceinline__ u16 f2bf(float f) {
    u32 x = __float_as_uint(f);
    u32 r = x + 0x7FFFu + ((x >> 16) & 1u);   // round-to-nearest-even
    return (u16)(r >> 16);
}
__device__ __forceinline__ float bf2f(u16 h) {
    return __uint_as_float(((u32)h) << 16);
}

// ---------------------------------------------------------------------------
// CDNA5 async global->LDS copy (ASYNCcnt-tracked), 16B per lane.
// ---------------------------------------------------------------------------
__device__ __forceinline__ void async_ld_b128(u32 ldsoff, const void* gptr) {
    asm volatile("global_load_async_to_lds_b128 %0, %1, off"
                 :: "v"(ldsoff), "v"(gptr)
                 : "memory");
}
__device__ __forceinline__ void wait_async_le2() {
#if __has_builtin(__builtin_amdgcn_s_wait_asynccnt)
    __builtin_amdgcn_s_wait_asynccnt(2);
#else
    asm volatile("s_wait_asynccnt 0x2" ::: "memory");
#endif
}
__device__ __forceinline__ void wait_async_le0() {
#if __has_builtin(__builtin_amdgcn_s_wait_asynccnt)
    __builtin_amdgcn_s_wait_asynccnt(0);
#else
    asm volatile("s_wait_asynccnt 0x0" ::: "memory");
#endif
}

// ---------------------------------------------------------------------------
// Generic bf16 WMMA GEMM:  C[M,N] (f32) = A[M,K] * B[N,K]^T  (+bias,+res,relu)
// A,B bf16 row-major, K contiguous. Block = 8 waves, tile 64(M) x 128(N).
// K loop staged as 64x64 LDS chunks, double buffered via ASYNC global->LDS
// loads (global_load_async_to_lds_b128 + s_wait_asynccnt). Per chunk:
// 2 half-chunks x (1 B frag + 4 A frags -> 4 independent wmma) = 8 v_wmma.
// K and M must be multiples of 64. Batched via grid.z.
// ---------------------------------------------------------------------------
__global__ __launch_bounds__(256)
void gemm_bf16_ex(const u16* __restrict__ A, const u16* __restrict__ B,
                  float* __restrict__ C,
                  int M, int N, int K, int lda, int ldb, int ldc,
                  long long sA, long long sB, long long sC,
                  const float* __restrict__ bias0,
                  const float* __restrict__ bias1, int nbias,
                  const float* __restrict__ res, int ldr, long long sRes,
                  int relu)
{
    __shared__ u16 lds_a[2][64][72];   // 64x64 chunk, stride 72 (16B-friendly)

    const int bz = blockIdx.z;
    const u16* Ab = A + (size_t)bz * sA;
    const u16* Bb = B + (size_t)bz * sB;
    float*     Cb = C + (size_t)bz * sC;
    const float* Rb = res ? (res + (size_t)bz * sRes) : (const float*)0;

    const int m0   = blockIdx.y * 64;
    const int wid  = threadIdx.x >> 5;
    const int lane = threadIdx.x & 31;
    const int n0   = blockIdx.x * 128 + wid * 16;
    const bool active = (n0 < N);

    const int klane = (lane >= 16) ? 8 : 0;
    const int arow  = lane & 15;
    const int brow  = n0 + (lane & 15);

    // staging coords: thread owns 16 bf16 (two 16B lines) of a 64x64 chunk
    const int sr = threadIdx.x >> 2;          // 0..63
    const int sc = (threadIdx.x & 3) * 8;     // 0,8,16,24
    const u16* aRow = Ab + (size_t)(m0 + sr) * lda;

    v8f acc[4] = { {}, {}, {}, {} };

    auto stageAsync = [&](int buf, int kc) {
        async_ld_b128((u32)(uintptr_t)&lds_a[buf][sr][sc],
                      (const void*)(aRow + kc + sc));
        async_ld_b128((u32)(uintptr_t)&lds_a[buf][sr][sc + 32],
                      (const void*)(aRow + kc + sc + 32));
    };

    stageAsync(0, 0);                          // chunk 0 in flight

    for (int kc = 0; kc < K; kc += 64) {
        const int buf = (kc >> 6) & 1;
        if (kc + 64 < K) {
            stageAsync(buf ^ 1, kc + 64);      // prefetch next chunk
            wait_async_le2();                  // current chunk's 2 loads done
        } else {
            wait_async_le0();
        }
        __syncthreads();
        if (active) {
            const u16* bRow = Bb + (size_t)brow * ldb + kc;
#pragma unroll
            for (int h = 0; h < 2; ++h) {
                BFrag b;
#pragma unroll
                for (int i = 0; i < 8; ++i) {
                    int ko = h * 32 + ((i >= 4) ? 16 : 0) + klane + ((i & 3) << 1);
                    b.u[i] = *(const u32*)(bRow + ko);
                }
                BFrag a[4];
#pragma unroll
                for (int mt = 0; mt < 4; ++mt) {
#pragma unroll
                    for (int i = 0; i < 8; ++i) {
                        int ko = h * 32 + ((i >= 4) ? 16 : 0) + klane + ((i & 3) << 1);
                        a[mt].u[i] = *(const u32*)&lds_a[buf][mt * 16 + arow][ko];
                    }
                }
#pragma unroll
                for (int mt = 0; mt < 4; ++mt) {
                    acc[mt] = __builtin_amdgcn_wmma_f32_16x16x32_bf16(
                        false, a[mt].v, false, b.v, (short)0, acc[mt], false, false);
                }
            }
        }
        __syncthreads();   // all reads of buf done before it is overwritten
    }

    if (active) {
        const int ncol = n0 + (lane & 15);
        float badd = 0.f;
        if (ncol < nbias) {
            if (bias0) badd += bias0[ncol];
            if (bias1) badd += bias1[ncol];
        }
#pragma unroll
        for (int mt = 0; mt < 4; ++mt) {
#pragma unroll
            for (int r = 0; r < 8; ++r) {
                int mrow = m0 + mt * 16 + r + ((lane >= 16) ? 8 : 0);
                float v = acc[mt][r] + badd;
                if (Rb) v += Rb[(size_t)mrow * ldr + ncol];
                if (relu) v = fmaxf(v, 0.f);
                Cb[(size_t)mrow * ldc + ncol] = v;
            }
        }
    }
}

// ---------------------------------------------------------------------------
// Fused embedding-gather + projection GEMM (VGPR staging: needs f32->bf16):
//   row r of A = bf16(embed_table[tokens[r]]), zero-padded EMB->K
//   C[M,N] f32 = A * Bw^T + bias.  K multiple of 64.
// ---------------------------------------------------------------------------
__global__ __launch_bounds__(256)
void gemm_embed_proj(const int* __restrict__ tokens,
                     const float* __restrict__ embed,
                     const u16* __restrict__ Bw, float* __restrict__ C,
                     int N, int K, int dvalid, int ldb, int ldc,
                     const float* __restrict__ bias, int nbias)
{
    __shared__ u16 lds_a[2][64][72];

    const int m0   = blockIdx.y * 64;
    const int wid  = threadIdx.x >> 5;
    const int lane = threadIdx.x & 31;
    const int n0   = blockIdx.x * 128 + wid * 16;
    const bool active = (n0 < N);

    const int klane = (lane >= 16) ? 8 : 0;
    const int arow  = lane & 15;
    const int brow  = n0 + (lane & 15);

    const int sr = threadIdx.x >> 2;
    const int sc = (threadIdx.x & 3) * 8;
    const int tok = tokens[m0 + sr];
    const float* aptr = embed + (size_t)tok * dvalid;

    float f[16];
    auto fetchA = [&](int kc) {
#pragma unroll
        for (int j = 0; j < 8; ++j) {
            int k0 = kc + sc + j;
            f[j]     = (k0      < dvalid) ? aptr[k0]      : 0.f;
            f[j + 8] = (k0 + 32 < dvalid) ? aptr[k0 + 32] : 0.f;
        }
    };
    auto packA = [&](int j0) {
        uint4 st;
        st.x = (u32)f2bf(f[j0 + 0]) | ((u32)f2bf(f[j0 + 1]) << 16);
        st.y = (u32)f2bf(f[j0 + 2]) | ((u32)f2bf(f[j0 + 3]) << 16);
        st.z = (u32)f2bf(f[j0 + 4]) | ((u32)f2bf(f[j0 + 5]) << 16);
        st.w = (u32)f2bf(f[j0 + 6]) | ((u32)f2bf(f[j0 + 7]) << 16);
        return st;
    };

    v8f acc[4] = { {}, {}, {}, {} };
    fetchA(0);

    for (int kc = 0; kc < K; kc += 64) {
        const int buf = (kc >> 6) & 1;
        *(uint4*)&lds_a[buf][sr][sc]      = packA(0);
        *(uint4*)&lds_a[buf][sr][sc + 32] = packA(8);
        if (kc + 64 < K) fetchA(kc + 64);
        __syncthreads();
        if (active) {
            const u16* bRow = Bw + (size_t)brow * ldb + kc;
#pragma unroll
            for (int h = 0; h < 2; ++h) {
                BFrag b;
#pragma unroll
                for (int i = 0; i < 8; ++i) {
                    int ko = h * 32 + ((i >= 4) ? 16 : 0) + klane + ((i & 3) << 1);
                    b.u[i] = *(const u32*)(bRow + ko);
                }
                BFrag a[4];
#pragma unroll
                for (int mt = 0; mt < 4; ++mt) {
#pragma unroll
                    for (int i = 0; i < 8; ++i) {
                        int ko = h * 32 + ((i >= 4) ? 16 : 0) + klane + ((i & 3) << 1);
                        a[mt].u[i] = *(const u32*)&lds_a[buf][mt * 16 + arow][ko];
                    }
                }
#pragma unroll
                for (int mt = 0; mt < 4; ++mt) {
                    acc[mt] = __builtin_amdgcn_wmma_f32_16x16x32_bf16(
                        false, a[mt].v, false, b.v, (short)0, acc[mt], false, false);
                }
            }
        }
        __syncthreads();
    }

    if (active) {
        const int ncol = n0 + (lane & 15);
        float badd = (ncol < nbias) ? bias[ncol] : 0.f;
#pragma unroll
        for (int mt = 0; mt < 4; ++mt) {
#pragma unroll
            for (int r = 0; r < 8; ++r) {
                int mrow = m0 + mt * 16 + r + ((lane >= 16) ? 8 : 0);
                C[(size_t)mrow * ldc + ncol] = acc[mt][r] + badd;
            }
        }
    }
}

// ---------------------------------------------------------------------------
// f32 -> bf16 with zero padding (weights)
// ---------------------------------------------------------------------------
__global__ void convert_pad_bf16(const float* __restrict__ W, u16* __restrict__ out,
                                 int rows, int cols, int orows, int ocols)
{
    int idx = blockIdx.x * blockDim.x + threadIdx.x;
    int total = orows * ocols;
    if (idx >= total) return;
    int r = idx / ocols, c = idx % ocols;
    float v = (r < rows && c < cols) ? W[(size_t)r * cols + c] : 0.f;
    out[idx] = f2bf(v);
}

// ---------------------------------------------------------------------------
// BatchNorm stats: one block per feature, reduce over rows
// ---------------------------------------------------------------------------
__global__ __launch_bounds__(256)
void bn_stats(const float* __restrict__ X, int rows, int ld,
              float* __restrict__ mean, float* __restrict__ rstd)
{
    int f = blockIdx.x;
    float s = 0.f, ss = 0.f;
    for (int r = threadIdx.x; r < rows; r += 256) {
        float v = X[(size_t)r * ld + f];
        s += v; ss += v * v;
    }
    __shared__ float sh[256], sh2[256];
    sh[threadIdx.x] = s; sh2[threadIdx.x] = ss;
    __syncthreads();
    for (int st = 128; st > 0; st >>= 1) {
        if (threadIdx.x < st) {
            sh[threadIdx.x]  += sh[threadIdx.x + st];
            sh2[threadIdx.x] += sh2[threadIdx.x + st];
        }
        __syncthreads();
    }
    if (threadIdx.x == 0) {
        float mu  = sh[0] / (float)rows;
        float var = sh2[0] / (float)rows - mu * mu;
        mean[f] = mu;
        rstd[f] = rsqrtf(fmaxf(var, 0.f) + 1e-5f);
    }
}

// BN apply + f32->bf16 (zero pads cols >= kvalid)
__global__ void bn_apply_bf16(const float* __restrict__ X, u16* __restrict__ out,
                              const float* __restrict__ mean,
                              const float* __restrict__ rstd,
                              const float* __restrict__ g,
                              const float* __restrict__ b,
                              int rows, int inld, int outld, int kvalid)
{
    int idx = blockIdx.x * blockDim.x + threadIdx.x;
    int total = rows * outld;
    if (idx >= total) return;
    int r = idx / outld, c = idx % outld;
    float v = 0.f;
    if (c < kvalid)
        v = (X[(size_t)r * inld + c] - mean[c]) * rstd[c] * g[c] + b[c];
    out[idx] = f2bf(v);
}

// ---------------------------------------------------------------------------
// LSTM elementwise gates. z:[B,4H] (i,f,g,o). Writes c and h (bf16) into the
// padded sequence buffer hsb[B,SPAD,H].
// ---------------------------------------------------------------------------
__global__ void lstm_gates(const float* __restrict__ z, float* __restrict__ c,
                           u16* __restrict__ hsb, int t, int first)
{
    int idx = blockIdx.x * blockDim.x + threadIdx.x;   // BATCH*HID
    if (idx >= BATCH * HID) return;
    int b = idx >> 9, j = idx & (HID - 1);
    const float* zr = z + (size_t)b * G4H;
    float zi = zr[j], zf = zr[HID + j], zg = zr[2 * HID + j], zo = zr[3 * HID + j];
    float cp = first ? 0.f : c[idx];
    float si = 1.f / (1.f + expf(-zi));
    float sf = 1.f / (1.f + expf(-zf));
    float so = 1.f / (1.f + expf(-zo));
    float cn = sf * cp + si * tanhf(zg);
    float h  = so * tanhf(cn);
    c[idx] = cn;
    hsb[(size_t)b * (SPAD * HID) + (size_t)t * HID + j] = f2bf(h);
}

// Row norms |h|^2 per (b,t) from bf16 sequence buffer
__global__ __launch_bounds__(128)
void row_norms(const u16* __restrict__ hsb, float* __restrict__ norms)
{
    int bs = blockIdx.x;                 // 0..BATCH*SEQ-1
    int b = bs / SEQ, t = bs % SEQ;
    const u16* row = hsb + (size_t)b * (SPAD * HID) + (size_t)t * HID;
    float s = 0.f;
    for (int j = threadIdx.x; j < HID; j += 128) {
        float v = bf2f(row[j]);
        s += v * v;
    }
    __shared__ float sh[128];
    sh[threadIdx.x] = s; __syncthreads();
    for (int st = 64; st > 0; st >>= 1) {
        if (threadIdx.x < st) sh[threadIdx.x] += sh[threadIdx.x + st];
        __syncthreads();
    }
    if (threadIdx.x == 0) norms[b * SPAD + t] = sh[0];
}

// Zero padded rows (t = 60..63) of both sequence buffers
__global__ void zero_hsb_pad(u16* __restrict__ h1, u16* __restrict__ h2)
{
    int idx = blockIdx.x * blockDim.x + threadIdx.x;
    if (idx >= BATCH * (SPAD - SEQ) * HID) return;
    int b   = idx / ((SPAD - SEQ) * HID);
    int rem = idx % ((SPAD - SEQ) * HID);
    int r   = SEQ + rem / HID;
    int j   = rem % HID;
    size_t off = (size_t)b * (SPAD * HID) + (size_t)r * HID + j;
    h1[off] = 0; h2[off] = 0;
}

// ---------------------------------------------------------------------------
// dist = sqrt(max(|h1|^2 + |h2|^2 - 2*gram, 0) + 1e-12); 4x4 min-pool;
// concat extra feats; emit bf16 MLP input row [MINP].
// ---------------------------------------------------------------------------
__global__ __launch_bounds__(256)
void dist_pool(const float* __restrict__ gram, const float* __restrict__ n1,
               const float* __restrict__ n2, const float* __restrict__ extra,
               u16* __restrict__ x0)
{
    int b = blockIdx.x;
    int tid = threadIdx.x;
    float val = 0.f;
    if (tid < 225) {
        int gi = tid / 15, gj = tid % 15;
        const float* gb = gram + (size_t)b * (SPAD * SPAD);
        float mn = 3.4e38f;
#pragma unroll
        for (int di = 0; di < 4; ++di) {
            int i = gi * 4 + di;
            float si = n1[b * SPAD + i];
#pragma unroll
            for (int dj = 0; dj < 4; ++dj) {
                int j = gj * 4 + dj;
                float sq = si + n2[b * SPAD + j] - 2.f * gb[i * SPAD + j];
                float d = sqrtf(fmaxf(sq, 0.f) + 1e-12f);
                mn = fminf(mn, d);
            }
        }
        val = mn;
    } else if (tid < 232) {
        val = extra[b * 7 + (tid - 225)];
    }
    x0[(size_t)b * MINP + tid] = f2bf(val);
}

// log_softmax over 2 classes
__global__ void logsoftmax2(const float* __restrict__ y, float* __restrict__ out)
{
    int b = threadIdx.x;                 // one block of 256
    float z0 = y[b * 16 + 0], z1 = y[b * 16 + 1];
    float m = fmaxf(z0, z1);
    float l = m + logf(expf(z0 - m) + expf(z1 - m));
    out[b * 2 + 0] = z0 - l;
    out[b * 2 + 1] = z1 - l;
}

// ---------------------------------------------------------------------------
// Host launch
// ---------------------------------------------------------------------------
extern "C" void kernel_launch(void* const* d_in, const int* in_sizes, int n_in,
                              void* d_out, int out_size, void* d_ws, size_t ws_size,
                              hipStream_t stream)
{
    const float* embed  = (const float*)d_in[0];
    const float* proj_W = (const float*)d_in[1];
    const float* proj_b = (const float*)d_in[2];
    const float* bn_g   = (const float*)d_in[3];
    const float* bn_b   = (const float*)d_in[4];
    const float* Wih    = (const float*)d_in[5];
    const float* Whh    = (const float*)d_in[6];
    const float* bih    = (const float*)d_in[7];
    const float* bhh    = (const float*)d_in[8];
    const float* W1     = (const float*)d_in[9];
    const float* b1     = (const float*)d_in[10];
    const float* g1     = (const float*)d_in[11];
    const float* be1    = (const float*)d_in[12];
    const float* W2     = (const float*)d_in[13];
    const float* b2     = (const float*)d_in[14];
    const float* g2     = (const float*)d_in[15];
    const float* be2    = (const float*)d_in[16];
    const float* W3     = (const float*)d_in[17];
    const float* b3     = (const float*)d_in[18];
    const float* extra  = (const float*)d_in[19];
    const int*   sent1  = (const int*)d_in[20];
    const int*   sent2  = (const int*)d_in[21];
    float* out = (float*)d_out;

    char* p = (char*)d_ws;
    auto alloc = [&](size_t bytes) -> void* {
        void* r = (void*)p;
        p += (bytes + 255) & ~(size_t)255;
        return r;
    };

    u16*   projWb = (u16*)  alloc((size_t)NPAD * DPAD * 2);
    u16*   Wihb   = (u16*)  alloc((size_t)G4H * DPAD * 2);
    u16*   Whhb   = (u16*)  alloc((size_t)G4H * HID * 2);
    u16*   W1b    = (u16*)  alloc((size_t)DM * MINP * 2);
    u16*   W2b    = (u16*)  alloc((size_t)DM * DM * 2);
    u16*   W3b    = (u16*)  alloc((size_t)16 * DM * 2);
    float* eproj  = (float*)alloc((size_t)SEQ * BATCH * NPAD * 4);
    u16*   xbf    = (u16*)  alloc((size_t)SEQ * BATCH * DPAD * 2);
    float* xWih   = (float*)alloc((size_t)SEQ * BATCH * G4H * 4);
    float* zbuf   = (float*)alloc((size_t)BATCH * G4H * 4);
    float* cbuf   = (float*)alloc((size_t)BATCH * HID * 4);
    u16*   hsb1   = (u16*)  alloc((size_t)BATCH * SPAD * HID * 2);
    u16*   hsb2   = (u16*)  alloc((size_t)BATCH * SPAD * HID * 2);
    float* nrm1   = (float*)alloc((size_t)BATCH * SPAD * 4);
    float* nrm2   = (float*)alloc((size_t)BATCH * SPAD * 4);
    float* gram   = (float*)alloc((size_t)BATCH * SPAD * SPAD * 4);
    u16*   x0     = (u16*)  alloc((size_t)BATCH * MINP * 2);
    float* y1     = (float*)alloc((size_t)BATCH * DM * 4);
    u16*   x1b    = (u16*)  alloc((size_t)BATCH * DM * 2);
    float* y2     = (float*)alloc((size_t)BATCH * DM * 4);
    u16*   x2b    = (u16*)  alloc((size_t)BATCH * DM * 2);
    float* y3     = (float*)alloc((size_t)BATCH * 16 * 4);
    float* meanb  = (float*)alloc(1024 * 4);
    float* rstdb  = (float*)alloc(1024 * 4);

    auto cvt = [&](const float* src, u16* dst, int r, int c, int orr, int oc) {
        int total = orr * oc;
        convert_pad_bf16<<<(total + 255) / 256, 256, 0, stream>>>(src, dst, r, c, orr, oc);
    };
    cvt(proj_W, projWb, EMB, EMB, NPAD, DPAD);
    cvt(Wih,    Wihb,   G4H, EMB, G4H,  DPAD);
    cvt(Whh,    Whhb,   G4H, HID, G4H,  HID);
    cvt(W1,     W1b,    DM,  MIN_, DM,  MINP);
    cvt(W2,     W2b,    DM,  DM,  DM,   DM);
    cvt(W3,     W3b,    2,   DM,  16,   DM);

    {
        int total = BATCH * (SPAD - SEQ) * HID;
        zero_hsb_pad<<<(total + 255) / 256, 256, 0, stream>>>(hsb1, hsb2);
    }

    const int* sents[2]  = { sent1, sent2 };
    u16*       hsbs[2]   = { hsb1, hsb2 };
    float*     nrms[2]   = { nrm1, nrm2 };
    const int M_SB = SEQ * BATCH;                         // 15360

    for (int s = 0; s < 2; ++s) {
        // 1) gather + projection GEMM -> eproj [M_SB, NPAD]
        {
            dim3 grid((NPAD + 127) / 128, M_SB / 64, 1);
            gemm_embed_proj<<<grid, 256, 0, stream>>>(
                sents[s], embed, projWb, eproj,
                NPAD, DPAD, EMB, DPAD, NPAD, proj_b, EMB);
        }
        // 2) BN over all rows, 300 features
        bn_stats<<<EMB, 256, 0, stream>>>(eproj, M_SB, NPAD, meanb, rstdb);
        {
            int total = M_SB * DPAD;
            bn_apply_bf16<<<(total + 255) / 256, 256, 0, stream>>>(
                eproj, xbf, meanb, rstdb, bn_g, bn_b, M_SB, NPAD, DPAD, EMB);
        }
        // 3) precompute x @ Wih^T + (bih + bhh) for all timesteps
        {
            dim3 grid(G4H / 128, M_SB / 64, 1);
            gemm_bf16_ex<<<grid, 256, 0, stream>>>(
                xbf, Wihb, xWih, M_SB, G4H, DPAD, DPAD, DPAD, G4H,
                0, 0, 0, bih, bhh, G4H, (const float*)0, 0, 0, 0);
        }
        // 4) recurrent loop
        for (int t = 0; t < SEQ; ++t) {
            const float* zsrc;
            if (t == 0) {
                zsrc = xWih;                               // h_{-1}=0
            } else {
                dim3 grid(G4H / 128, BATCH / 64, 1);
                gemm_bf16_ex<<<grid, 256, 0, stream>>>(
                    hsbs[s] + (size_t)(t - 1) * HID,       // A row b at b*SPAD*HID
                    Whhb, zbuf,
                    BATCH, G4H, HID, SPAD * HID, HID, G4H,
                    0, 0, 0, (const float*)0, (const float*)0, 0,
                    xWih + (size_t)t * BATCH * G4H, G4H, 0, 0);
                zsrc = zbuf;
            }
            int total = BATCH * HID;
            lstm_gates<<<(total + 255) / 256, 256, 0, stream>>>(
                zsrc, cbuf, hsbs[s], t, (t == 0) ? 1 : 0);
        }
        // 5) per-row squared norms
        row_norms<<<BATCH * SEQ, 128, 0, stream>>>(hsbs[s], nrms[s]);
    }

    // 6) batched Gram: gram[b] = h1[b] (64x512) * h2[b]^T
    {
        dim3 grid(1, SPAD / 64, BATCH);
        gemm_bf16_ex<<<grid, 256, 0, stream>>>(
            hsb1, hsb2, gram, SPAD, SPAD, HID, HID, HID, SPAD,
            (long long)SPAD * HID, (long long)SPAD * HID, (long long)SPAD * SPAD,
            (const float*)0, (const float*)0, 0, (const float*)0, 0, 0, 0);
    }
    // 7) distances + 4x4 min-pool + concat extra -> bf16 MLP input
    dist_pool<<<BATCH, 256, 0, stream>>>(gram, nrm1, nrm2, extra, x0);

    // 8) MLP layer 1: relu(x0 W1^T + b1) -> BN -> bf16
    {
        dim3 grid(DM / 128, BATCH / 64, 1);
        gemm_bf16_ex<<<grid, 256, 0, stream>>>(
            x0, W1b, y1, BATCH, DM, MINP, MINP, MINP, DM,
            0, 0, 0, b1, (const float*)0, DM, (const float*)0, 0, 0, 1);
        bn_stats<<<DM, 256, 0, stream>>>(y1, BATCH, DM, meanb, rstdb);
        int total = BATCH * DM;
        bn_apply_bf16<<<(total + 255) / 256, 256, 0, stream>>>(
            y1, x1b, meanb, rstdb, g1, be1, BATCH, DM, DM, DM);
    }
    // 9) MLP layer 2
    {
        dim3 grid(DM / 128, BATCH / 64, 1);
        gemm_bf16_ex<<<grid, 256, 0, stream>>>(
            x1b, W2b, y2, BATCH, DM, DM, DM, DM, DM,
            0, 0, 0, b2, (const float*)0, DM, (const float*)0, 0, 0, 1);
        bn_stats<<<DM, 256, 0, stream>>>(y2, BATCH, DM, meanb, rstdb);
        int total = BATCH * DM;
        bn_apply_bf16<<<(total + 255) / 256, 256, 0, stream>>>(
            y2, x2b, meanb, rstdb, g2, be2, BATCH, DM, DM, DM);
    }
    // 10) output layer (N padded 2->16, bias valid only for n<2)
    {
        dim3 grid(1, BATCH / 64, 1);
        gemm_bf16_ex<<<grid, 256, 0, stream>>>(
            x2b, W3b, y3, BATCH, 16, DM, DM, DM, 16,
            0, 0, 0, b3, (const float*)0, 2, (const float*)0, 0, 0, 0);
    }
    // 11) log_softmax -> d_out [256,2]
    logsoftmax2<<<1, 256, 0, stream>>>(y3, out);

    (void)in_sizes; (void)n_in; (void)out_size; (void)ws_size;
}
